// QMixer_gat_78469052498389
// MI455X (gfx1250) — matrix operand order repaired and checked by reading, not compiled, not run on hardware
//
#include <hip/hip_runtime.h>
#include <hip/hip_bf16.h>

// ---------------- problem constants ----------------
#define BT        262144      // BS * T
#define SPB       32          // samples per block (2 WMMA M-tiles)
#define NTHREADS  256         // 8 waves (wave32)
#define K_DIM     512         // LAYER_IN
#define N_COLS    352         // 256 (hw1) + 32 (hb1) + 32 (hwf) + 32 (v1)
#define N_PAD     384         // padded to 24 N-tiles so 8 waves x 6 tiles, branchless
#define S_STRIDE  520         // bf16 units per s-row in LDS (512 + 8 pad)
#define O_STRIDE  388         // f32 units per out-row in LDS (384 + 4 pad)

typedef __attribute__((ext_vector_type(16))) __bf16 v16bf;
typedef __attribute__((ext_vector_type(8)))  float  v8f;

union AFrag { v16bf v; unsigned int u[8]; };
union BFrag { v16bf v; uint4 q[2]; };

__device__ __forceinline__ unsigned short f2bf(float f) {
    union { float f; unsigned int u; } c; c.f = f;
    unsigned int u = c.u;
    u += 0x7FFFu + ((u >> 16) & 1u);           // round-to-nearest-even
    return (unsigned short)(u >> 16);
}
__device__ __forceinline__ float eluf(float x) { return x > 0.f ? x : expm1f(x); }

// ---------------- kernel 0: weights -> bf16 B in WMMA-fragment order (N padded to 384) ----------------
// Tile (nt,kt) = contiguous 1KB: piece p (0/1) * 512B, lane * 16B, elem * 2B.
// Lane l of a wave holds column n = nt*16 + (l&15), K = kt*32 + (l>>4)*16 + p*8 + e.
__global__ void wconvert(const float* __restrict__ hw1, const float* __restrict__ hb1,
                         const float* __restrict__ hwf, const float* __restrict__ v1,
                         unsigned short* __restrict__ B) {
    int idx = blockIdx.x * blockDim.x + threadIdx.x;
    if (idx >= N_PAD * K_DIM) return;
    int n = idx / K_DIM, k = idx - n * K_DIM;
    float f = 0.f;                                  // zero-fill pad tiles (deterministic)
    if (n < 256)      f = hw1[k * 256 + n];
    else if (n < 288) f = hb1[k * 32 + (n - 256)];
    else if (n < 320) f = hwf[k * 32 + (n - 288)];
    else if (n < 352) f = v1 [k * 32 + (n - 320)];
    int nt = n >> 4, ln = n & 15;
    int kt = k >> 5, kk = k & 31;
    int h  = kk >> 4;            // lane half
    int p  = (kk >> 3) & 1;      // 16B piece within lane
    int e  = kk & 7;             // element within piece
    int lanei = h * 16 + ln;
    size_t off = ((size_t)(nt * 16 + kt) << 9) + (p << 8) + lanei * 8 + e;  // u16 units
    B[off] = f2bf(f);
}

// ---------------- fused kernel: GAT -> hypernet GEMM (WMMA bf16) -> QMIX mixing ----------------
__launch_bounds__(NTHREADS)
__global__ void qmix_fused(const float* __restrict__ agent_qs,
                           const float* __restrict__ states,
                           const float* __restrict__ Wh_, const float* __restrict__ Ah_,
                           const float* __restrict__ Wo_, const float* __restrict__ Ao_,
                           const float* __restrict__ hw1_b, const float* __restrict__ hb1_b,
                           const float* __restrict__ hwf_b, const float* __restrict__ v1_b,
                           const float* __restrict__ v2_w,  const float* __restrict__ v2_b,
                           const unsigned short* __restrict__ Bw,
                           float* __restrict__ out)
{
    __shared__ float gatW[256];     // W_heads [4][16][4]
    __shared__ float gatA[32];      // a_heads [4][8]
    __shared__ float gatWo[512];    // W_out   [16][32]
    __shared__ float gatAo[64];     // a_out   [64]
    __shared__ float bia[385];      // [hw1_b|hb1_b|hwf_b|v1_b|v2_w|v2_b]
    __shared__ float qs_lds[SPB * 8];
    __shared__ __align__(16) unsigned short s_lds[SPB * S_STRIDE];   // bf16 s rows
    __shared__ __align__(16) float out_lds[SPB * O_STRIDE];          // GEMM results

    const int tid  = threadIdx.x;
    const int lane = tid & 31;
    const int wave = tid >> 5;
    const int half = lane >> 4;       // which 16-lane half
    const int node = lane & 15;
    const int bl   = half << 4;       // shuffle base lane for this half
    const long long blockBase = (long long)blockIdx.x * SPB;

    // ---- stage shared parameters ----
    for (int i = tid; i < 256; i += NTHREADS) gatW[i] = Wh_[i];
    if (tid < 32)  gatA[tid]  = Ah_[tid];
    for (int i = tid; i < 512; i += NTHREADS) gatWo[i] = Wo_[i];
    if (tid < 64)  gatAo[tid] = Ao_[tid];
    for (int i = tid; i < 256; i += NTHREADS) bia[i] = hw1_b[i];
    if (tid < 32) {
        bia[256 + tid] = hb1_b[tid];
        bia[288 + tid] = hwf_b[tid];
        bia[320 + tid] = v1_b[tid];
        bia[352 + tid] = v2_w[tid];
    }
    if (tid == 0) bia[384] = v2_b[0];
    qs_lds[tid] = agent_qs[blockBase * 8 + tid];
    __syncthreads();

    // ================= Phase 1: GAT (2 samples per wave at a time, lane = node) =================
    for (int p = 0; p < 2; ++p) {
        const int ls = wave * 4 + p * 2 + half;          // local sample 0..31
        const long long g = blockBase + ls;
        const float* xp = states + g * 256 + node * 16;
        float xf[16];
        #pragma unroll
        for (int i = 0; i < 4; ++i) {
            float4 v4 = ((const float4*)xp)[i];
            xf[4*i] = v4.x; xf[4*i+1] = v4.y; xf[4*i+2] = v4.z; xf[4*i+3] = v4.w;
        }
        // Wh for all 4 heads packed: wh[h*4+k]
        float wh[16];
        #pragma unroll
        for (int i = 0; i < 16; ++i) wh[i] = 0.f;
        #pragma unroll
        for (int f = 0; f < 16; ++f) {
            float xv = xf[f];
            #pragma unroll
            for (int h = 0; h < 4; ++h)
                #pragma unroll
                for (int k = 0; k < 4; ++k)
                    wh[h*4+k] = fmaf(xv, gatW[h*64 + f*4 + k], wh[h*4+k]);
        }
        float hcat[16];
        #pragma unroll
        for (int h = 0; h < 4; ++h) {
            float e1 = 0.f, e2 = 0.f;
            #pragma unroll
            for (int k = 0; k < 4; ++k) {
                e1 = fmaf(wh[h*4+k], gatA[h*8 + k],     e1);
                e2 = fmaf(wh[h*4+k], gatA[h*8 + 4 + k], e2);
            }
            float m = -3.0e38f;
            for (int j = 0; j < 16; ++j) {
                float s = e1 + __shfl(e2, bl + j, 32);
                float l = s > 0.f ? s : 0.2f * s;
                m = fmaxf(m, l);
            }
            float Z = 0.f, hk0 = 0.f, hk1 = 0.f, hk2 = 0.f, hk3 = 0.f;
            for (int j = 0; j < 16; ++j) {
                float s = e1 + __shfl(e2, bl + j, 32);
                float l = s > 0.f ? s : 0.2f * s;
                float a = expf(l - m);
                Z += a;
                hk0 = fmaf(a, __shfl(wh[h*4+0], bl + j, 32), hk0);
                hk1 = fmaf(a, __shfl(wh[h*4+1], bl + j, 32), hk1);
                hk2 = fmaf(a, __shfl(wh[h*4+2], bl + j, 32), hk2);
                hk3 = fmaf(a, __shfl(wh[h*4+3], bl + j, 32), hk3);
            }
            float iz = 1.f / Z;
            hcat[h*4+0] = eluf(hk0 * iz);
            hcat[h*4+1] = eluf(hk1 * iz);
            hcat[h*4+2] = eluf(hk2 * iz);
            hcat[h*4+3] = eluf(hk3 * iz);
        }
        // output attention layer: [16x16] @ [16x32]
        float wh2[32];
        #pragma unroll
        for (int c = 0; c < 32; ++c) wh2[c] = 0.f;
        #pragma unroll
        for (int f = 0; f < 16; ++f) {
            float hv = hcat[f];
            #pragma unroll
            for (int c = 0; c < 32; ++c)
                wh2[c] = fmaf(hv, gatWo[f*32 + c], wh2[c]);
        }
        float e1 = 0.f, e2 = 0.f;
        #pragma unroll
        for (int c = 0; c < 32; ++c) {
            e1 = fmaf(wh2[c], gatAo[c],      e1);
            e2 = fmaf(wh2[c], gatAo[32 + c], e2);
        }
        float m = -3.0e38f;
        for (int j = 0; j < 16; ++j) {
            float s = e1 + __shfl(e2, bl + j, 32);
            float l = s > 0.f ? s : 0.2f * s;
            m = fmaxf(m, l);
        }
        float Z = 0.f;
        float ho[32];
        #pragma unroll
        for (int c = 0; c < 32; ++c) ho[c] = 0.f;
        for (int j = 0; j < 16; ++j) {
            float s = e1 + __shfl(e2, bl + j, 32);
            float l = s > 0.f ? s : 0.2f * s;
            float a = expf(l - m);
            Z += a;
            #pragma unroll
            for (int c = 0; c < 32; ++c)
                ho[c] = fmaf(a, __shfl(wh2[c], bl + j, 32), ho[c]);
        }
        float iz = 1.f / Z;
        // s row (bf16): s[node*32 + c] = elu(gat_out)
        unsigned int* srow = (unsigned int*)&s_lds[ls * S_STRIDE + node * 32];
        #pragma unroll
        for (int c = 0; c < 32; c += 2) {
            unsigned int lo = f2bf(eluf(ho[c]     * iz));
            unsigned int hi = f2bf(eluf(ho[c + 1] * iz));
            srow[c >> 1] = lo | (hi << 16);
        }
    }
    __syncthreads();

    // ===== Phase 2: hypernet GEMM [32 x 512] x [512 x 384(pad)]  (bf16 WMMA, branchless) =====
    {
        const int mt   = wave >> 2;            // 0..1  (M tile)
        const int wcol = wave & 3;             // 0..3  (N tile group, exactly 6 tiles each)
        const int row  = mt * 16 + (lane & 15);
        const unsigned int* sbase = (const unsigned int*)&s_lds[row * S_STRIDE];

        const v8f vzero = {0.f,0.f,0.f,0.f,0.f,0.f,0.f,0.f};
        v8f acc[6];
        #pragma unroll
        for (int i = 0; i < 6; ++i) acc[i] = vzero;

        #pragma unroll
        for (int kt = 0; kt < 16; ++kt) {
            // A fragment: lanes 0-15 hold K 0-15, lanes 16-31 hold K 16-31
            AFrag a;
            #pragma unroll
            for (int p = 0; p < 8; ++p) {
                int K = 2 * p + (half << 4);
                a.u[p] = sbase[(kt * 32 + K) >> 1];
            }
            #pragma unroll
            for (int i = 0; i < 6; ++i) {
                // B fragment in swizzled order: contiguous 1KB tile per (nt,kt);
                // lane l reads uint4 at tile*64 + l and +32 (fully coalesced)
                const int tile = (wcol * 6 + i) * 16 + kt;
                const uint4* bp = (const uint4*)Bw + ((size_t)tile << 6) + lane;
                BFrag b;
                b.q[0] = bp[0];
                b.q[1] = bp[32];
                acc[i] = __builtin_amdgcn_wmma_f32_16x16x32_bf16(
                             false, a.v, false, b.v, (short)0, acc[i], false, false);
            }
        }
        #pragma unroll
        for (int i = 0; i < 6; ++i) {
            const int nt = wcol * 6 + i;
            #pragma unroll
            for (int r = 0; r < 8; ++r) {
                int mrow = mt * 16 + r + (half << 3);
                out_lds[mrow * O_STRIDE + nt * 16 + (lane & 15)] = acc[i][r];
            }
        }
    }
    __syncthreads();

    // ================= Phase 3: QMIX mixing (1 wave per 4 samples, lane = embed index) =================
    for (int q = 0; q < 4; ++q) {
        const int s = wave * 4 + q;
        const float* o  = &out_lds[s * O_STRIDE];
        const float* qs = &qs_lds[s * 8];
        float acc = 0.f;
        #pragma unroll
        for (int a = 0; a < 8; ++a)
            acc = fmaf(qs[a], fabsf(o[a*32 + lane] + bia[a*32 + lane]), acc);
        float hid = eluf(acc + o[256 + lane] + bia[256 + lane]);   // elu(qs.w1 + b1)
        float wf  = fabsf(o[288 + lane] + bia[288 + lane]);        // |w_final|
        float vh  = fmaxf(o[320 + lane] + bia[320 + lane], 0.f);   // relu(v hidden)
        float t   = hid * wf + vh * bia[352 + lane];               // hidden.w_final + vh.v2_w
        #pragma unroll
        for (int off = 16; off > 0; off >>= 1)
            t += __shfl_xor(t, off, 32);
        if (lane == 0)
            out[blockBase + s] = t + bia[384];                     // + v2_b
    }
}

extern "C" void kernel_launch(void* const* d_in, const int* in_sizes, int n_in,
                              void* d_out, int out_size, void* d_ws, size_t ws_size,
                              hipStream_t stream) {
    const float* agent_qs = (const float*)d_in[0];
    const float* states   = (const float*)d_in[1];
    const float* W_heads  = (const float*)d_in[2];
    const float* a_heads  = (const float*)d_in[3];
    const float* W_out    = (const float*)d_in[4];
    const float* a_out    = (const float*)d_in[5];
    const float* hw1_w    = (const float*)d_in[6];
    const float* hw1_b    = (const float*)d_in[7];
    const float* hb1_w    = (const float*)d_in[8];
    const float* hb1_b    = (const float*)d_in[9];
    const float* hwf_w    = (const float*)d_in[10];
    const float* hwf_b    = (const float*)d_in[11];
    const float* v1_w     = (const float*)d_in[12];
    const float* v1_b     = (const float*)d_in[13];
    const float* v2_w     = (const float*)d_in[14];
    const float* v2_b     = (const float*)d_in[15];
    unsigned short* Bw = (unsigned short*)d_ws;     // needs 384*512*2 = 393216 bytes

    wconvert<<<(N_PAD * K_DIM + 255) / 256, 256, 0, stream>>>(hw1_w, hb1_w, hwf_w, v1_w, Bw);
    qmix_fused<<<BT / SPB, NTHREADS, 0, stream>>>(
        agent_qs, states, W_heads, a_heads, W_out, a_out,
        hw1_b, hb1_b, hwf_b, v1_b, v2_w, v2_b, Bw, (float*)d_out);
}
